// Ernie4_5_VL_MoeSparseMoeBlock_10831907520596
// MI455X (gfx1250) — matrix operand order repaired and checked by reading, not compiled, not run on hardware
//
#include <hip/hip_runtime.h>

// ---------------------------------------------------------------------------
// Ernie4.5 MoE sparse block for MI455X (gfx1250, wave32, WMMA + TDM).
// fp32 router/softmax/top-k/SiLU/accumulators; bf16 WMMA operands.
// Activation tiles staged to LDS via the Tensor Data Mover; weight tiles
// converted fp32->bf16 with packed cvt + dword LDS stores.
// ---------------------------------------------------------------------------

#define T_TOK 2048
#define H_DIM 2048
#define I_DIM 1024
#define E_NUM 8
#define KI    (E_NUM * I_DIM)   // 8192

typedef __attribute__((ext_vector_type(16))) __bf16       v16bf;
typedef __attribute__((ext_vector_type(8)))  float        v8f;
typedef __attribute__((ext_vector_type(4)))  unsigned int v4u;
typedef __attribute__((ext_vector_type(8)))  int          v8i;
typedef __attribute__((ext_vector_type(4)))  int          v4i;

union Frag16 { v16bf v; unsigned int d[8]; };

__device__ __forceinline__ unsigned int pack_bf16(float lo, float hi) {
    union { __bf16 h; unsigned short u; } a, b;
    a.h = (__bf16)lo; b.h = (__bf16)hi;
    return ((unsigned int)b.u << 16) | (unsigned int)a.u;
}

#if __has_builtin(__builtin_amdgcn_tensor_load_to_lds)
#define HAVE_TDM 1
#else
#define HAVE_TDM 0
#endif

__device__ __forceinline__ void tdm_wait0() {
#if __has_builtin(__builtin_amdgcn_s_wait_tensorcnt)
    __builtin_amdgcn_s_wait_tensorcnt(0);
#else
    asm volatile("s_wait_tensorcnt 0x0" ::: "memory");
#endif
}

#if HAVE_TDM
// 2D bf16 tile load Global->LDS via TDM. D# per CDNA5 ISA:
// group0: count=1 | lds_addr | global_addr[56:0] | type=2
// group1: data_size=1(2B) | tensor_dim0/1 | tile_dim0/1 | tensor_dim0_stride
__device__ __forceinline__ void tdm_load_tile_bf16(
    unsigned lds_off, const __bf16* gptr,
    unsigned tensor_w, unsigned tensor_h, unsigned stride_elems,
    unsigned tile_w, unsigned tile_h)
{
    unsigned long long ga = (unsigned long long)(const void*)gptr;
    v4u g0 = { 1u,                                   // count=1, user mode
               lds_off,                              // lds_addr (bytes)
               (unsigned)ga,                         // global_addr[31:0]
               (unsigned)((ga >> 32) & 0x01FFFFFFull) | 0x80000000u }; // [56:32]|type=2
    v8i g1 = { (int)(1u << 16),                                          // data_size=2B
               (int)((tensor_w & 0xFFFFu) << 16),                        // dim0[15:0]
               (int)(((tensor_w >> 16) & 0xFFFFu) |
                     ((tensor_h & 0xFFFFu) << 16)),                      // dim0[31:16]|dim1[15:0]
               (int)(((tensor_h >> 16) & 0xFFFFu) | (tile_w << 16)),     // dim1[31:16]|tile0
               (int)(tile_h & 0xFFFFu),                                  // tile1 (tile2=0)
               (int)stride_elems,                                        // dim0_stride[31:0]
               0, 0 };
    v4i z4 = {0, 0, 0, 0};
#if defined(__clang_major__) && (__clang_major__ >= 23)
    v8i z8 = {0, 0, 0, 0, 0, 0, 0, 0};
    __builtin_amdgcn_tensor_load_to_lds(g0, g1, z4, z4, z8, 0);
#else
    __builtin_amdgcn_tensor_load_to_lds(g0, g1, z4, z4, 0);
#endif
}
#endif

// ---------------------------------------------------------------------------
// Kernel 0: one-shot fp32 -> bf16 conversion of hidden_states (makes the
// activation tiles TDM-movable and avoids re-converting X 128x per tile).
// ---------------------------------------------------------------------------
__global__ __launch_bounds__(256) void moe_cvt_x_kernel(
    const float* __restrict__ x, __bf16* __restrict__ xbf)
{
    const size_t i = ((size_t)blockIdx.x * 256 + threadIdx.x) * 8;
    unsigned int* dst = (unsigned int*)(xbf + i);
#pragma unroll
    for (int j = 0; j < 8; j += 2)
        dst[j >> 1] = pack_bf16(x[i + j], x[i + j + 1]);
}

// ---------------------------------------------------------------------------
// Kernel 1: router. One wave32 per token, fp32 throughout.
// ---------------------------------------------------------------------------
__global__ __launch_bounds__(256) void moe_router_kernel(
    const float* __restrict__ x, const float* __restrict__ gw,
    const float* __restrict__ bias, float* __restrict__ logits_out,
    float* __restrict__ combine)
{
    const int wave = threadIdx.x >> 5;
    const int lane = threadIdx.x & 31;
    const int t    = blockIdx.x * 8 + wave;

    float acc[E_NUM];
#pragma unroll
    for (int e = 0; e < E_NUM; ++e) acc[e] = 0.0f;

    const float* xr = x + (size_t)t * H_DIM;
    for (int k = lane; k < H_DIM; k += 32) {
        const float xv = xr[k];
#pragma unroll
        for (int e = 0; e < E_NUM; ++e) acc[e] += xv * gw[e * H_DIM + k];
    }
#pragma unroll
    for (int off = 16; off > 0; off >>= 1) {
#pragma unroll
        for (int e = 0; e < E_NUM; ++e) acc[e] += __shfl_xor(acc[e], off, 32);
    }

    float mx = acc[0];
#pragma unroll
    for (int e = 1; e < E_NUM; ++e) mx = fmaxf(mx, acc[e]);
    float ex[E_NUM], sum = 0.0f;
#pragma unroll
    for (int e = 0; e < E_NUM; ++e) { ex[e] = __expf(acc[e] - mx); sum += ex[e]; }
    const float rs = __builtin_amdgcn_rcpf(sum);
    float probs[E_NUM], corr[E_NUM];
#pragma unroll
    for (int e = 0; e < E_NUM; ++e) {
        probs[e] = ex[e] * rs;
        corr[e]  = probs[e] + bias[e];
    }
    int i0 = 0; float b0 = corr[0];
#pragma unroll
    for (int e = 1; e < E_NUM; ++e) if (corr[e] > b0) { b0 = corr[e]; i0 = e; }
    int i1 = -1; float b1 = -3.4e38f;
#pragma unroll
    for (int e = 0; e < E_NUM; ++e)
        if (e != i0 && corr[e] > b1) { b1 = corr[e]; i1 = e; }
    float w0 = probs[i0], w1 = probs[i1];
    const float s = fmaxf(w0 + w1, 1e-12f);
    w0 /= s; w1 /= s;

    if (lane < E_NUM) {
        logits_out[(size_t)t * E_NUM + lane] = acc[lane];
        combine[(size_t)t * E_NUM + lane] =
            (lane == i0) ? w0 : (lane == i1) ? w1 : 0.0f;
    }
}

// ---------------------------------------------------------------------------
// Kernel 2: gate_up GEMM + SiLU + combine scaling. Tile 128 tokens x 64 i.
// 8 waves; each wave owns a 2x2 grid of 16x16 positions for BOTH gate and up
// (8 WMMA / wave / K-step). X tile via TDM; W tile via packed cvt stores.
// ---------------------------------------------------------------------------
__global__ __launch_bounds__(256) void moe_gateup_kernel(
    const __bf16* __restrict__ xbf,     // [T,H] bf16
    const float* __restrict__ gup,      // [E,H,2I] fp32
    const float* __restrict__ combine,  // [T,E]
    __bf16* __restrict__ hs)            // [T, E*I]
{
    __shared__ __bf16       Xs[128 * 32];   // [row][k]  (TDM row-major tile)
    __shared__ unsigned int WsU[128 * 16];  // [col][k-pair dword], K-transposed

    const int e  = blockIdx.z;
    const int t0 = blockIdx.y * 128;
    const int c0 = blockIdx.x * 64;
    const float* W = gup + (size_t)e * H_DIM * (2 * I_DIM);

    const int tid    = threadIdx.x;
    const int wv     = tid >> 5;
    const int lane   = tid & 31;
    const int mp     = wv >> 1;          // row-strip pair {2mp, 2mp+1}
    const int np     = wv & 1;           // n-tile pair {2np, 2np+1}
    const int laneN  = lane & 15;
    const int laneHi = lane >> 4;

    v8f accg[2][2] = {};
    v8f accu[2][2] = {};

#if HAVE_TDM
    const unsigned ldsXs = (unsigned)(unsigned long long)(const void*)Xs;
#endif

    for (int k0 = 0; k0 < H_DIM; k0 += 32) {
        __syncthreads();
#if HAVE_TDM
        if (wv == 0)
            tdm_load_tile_bf16(ldsXs, xbf + (size_t)t0 * H_DIM + k0,
                               H_DIM, T_TOK, H_DIM, 32, 128);
#else
        {   // fallback: cooperative bf16 copy, 8 dwords/thread
            const int row = tid >> 1, kd = (tid & 1) * 8;
            const unsigned int* src =
                (const unsigned int*)(xbf + (size_t)(t0 + row) * H_DIM + k0) + kd;
            unsigned int* dst = (unsigned int*)Xs + row * 16 + kd;
#pragma unroll
            for (int j = 0; j < 8; ++j) dst[j] = src[j];
        }
#endif
        {   // weights: 32k x 128 cols (64 gate + 64 up), k-pair packed stores
            const int kp  = tid >> 4;            // 0..15 -> k = 2kp
            const int cc0 = (tid & 15) * 8;      // 0..120, 8|64 -> no straddle
            const int gcol = (cc0 < 64) ? (c0 + cc0) : (I_DIM + c0 + (cc0 - 64));
            const float* s0 = W + (size_t)(k0 + 2 * kp) * (2 * I_DIM) + gcol;
            const float* s1 = s0 + 2 * I_DIM;
            if (k0 + 32 < H_DIM)
                __builtin_prefetch(s0 + 32 * (2 * I_DIM), 0, 1);
#pragma unroll
            for (int j = 0; j < 8; ++j)
                WsU[(cc0 + j) * 16 + kp] = pack_bf16(s0[j], s1[j]);
        }
#if HAVE_TDM
        if (wv == 0) tdm_wait0();
#endif
        __syncthreads();

        Frag16 a[2];
#pragma unroll
        for (int p = 0; p < 2; ++p) {
            const int m = (2 * mp + p) * 16 + laneN;
            const unsigned int* base = (const unsigned int*)&Xs[m * 32];
            const int g = laneHi * 8;
#pragma unroll
            for (int d = 0; d < 4; ++d) a[p].d[d]     = base[g + d];
#pragma unroll
            for (int d = 0; d < 4; ++d) a[p].d[4 + d] = base[4 + g + d];
        }
#pragma unroll
        for (int q = 0; q < 2; ++q) {
            const int n = (2 * np + q) * 16 + laneN;
            Frag16 bg, bu;
            const unsigned int* bgp = &WsU[n * 16];
            const unsigned int* bup = &WsU[(64 + n) * 16];
            const int g = laneHi * 8;
#pragma unroll
            for (int d = 0; d < 8; ++d) { bg.d[d] = bgp[g + d]; bu.d[d] = bup[g + d]; }
#pragma unroll
            for (int p = 0; p < 2; ++p) {
                accg[p][q] = __builtin_amdgcn_wmma_f32_16x16x32_bf16(
                    false, a[p].v, false, bg.v, (short)0, accg[p][q], false, false);
                accu[p][q] = __builtin_amdgcn_wmma_f32_16x16x32_bf16(
                    false, a[p].v, false, bu.v, (short)0, accu[p][q], false, false);
            }
        }
    }

    // epilogue: h = silu(gate) * up * combine[t,e]  (rcp instead of divide)
#pragma unroll
    for (int p = 0; p < 2; ++p) {
#pragma unroll
        for (int q = 0; q < 2; ++q) {
            const int col = e * I_DIM + c0 + (2 * np + q) * 16 + laneN;
#pragma unroll
            for (int r = 0; r < 8; ++r) {
                const int t  = t0 + (2 * mp + p) * 16 + laneHi * 8 + r;
                const float gt = accg[p][q][r];
                const float up = accu[p][q][r];
                const float h  = gt * __builtin_amdgcn_rcpf(1.0f + __expf(-gt)) *
                                 up * combine[t * E_NUM + e];
                hs[(size_t)t * KI + col] = (__bf16)h;
            }
        }
    }
}

// ---------------------------------------------------------------------------
// Kernel 3: down projection. out[T,H] = HS[T,8192](bf16) @ Wd[8192,H](fp32).
// Tile 128 tokens x 64 h; HS tile via TDM; Wd tile via packed cvt stores.
// ---------------------------------------------------------------------------
__global__ __launch_bounds__(256) void moe_down_kernel(
    const __bf16* __restrict__ hs,      // [T, 8192]
    const float* __restrict__ wd,       // [8192, H]
    float* __restrict__ out)            // [T, H]
{
    __shared__ __bf16       As[128 * 32];
    __shared__ unsigned int BsU[64 * 16];

    const int t0 = blockIdx.y * 128;
    const int h0 = blockIdx.x * 64;

    const int tid    = threadIdx.x;
    const int wv     = tid >> 5;
    const int lane   = tid & 31;
    const int mp     = wv >> 1;
    const int np     = wv & 1;
    const int laneN  = lane & 15;
    const int laneHi = lane >> 4;

    v8f acc[2][2] = {};

#if HAVE_TDM
    const unsigned ldsAs = (unsigned)(unsigned long long)(const void*)As;
#endif

    for (int k0 = 0; k0 < KI; k0 += 32) {
        __syncthreads();
#if HAVE_TDM
        if (wv == 0)
            tdm_load_tile_bf16(ldsAs, hs + (size_t)t0 * KI + k0,
                               KI, T_TOK, KI, 32, 128);
#else
        {
            const int row = tid >> 1, kd = (tid & 1) * 8;
            const unsigned int* src =
                (const unsigned int*)(hs + (size_t)(t0 + row) * KI + k0) + kd;
            unsigned int* dst = (unsigned int*)As + row * 16 + kd;
#pragma unroll
            for (int j = 0; j < 8; ++j) dst[j] = src[j];
        }
#endif
        {   // Wd tile: 32k x 64 cols, k-pair packed stores (4 cols/thread)
            const int kp  = tid >> 4;
            const int cc0 = (tid & 15) * 4;
            const float* s0 = wd + (size_t)(k0 + 2 * kp) * H_DIM + h0 + cc0;
            const float* s1 = s0 + H_DIM;
            if (k0 + 32 < KI)
                __builtin_prefetch(s0 + 32 * H_DIM, 0, 1);
#pragma unroll
            for (int j = 0; j < 4; ++j)
                BsU[(cc0 + j) * 16 + kp] = pack_bf16(s0[j], s1[j]);
        }
#if HAVE_TDM
        if (wv == 0) tdm_wait0();
#endif
        __syncthreads();

        Frag16 a[2];
#pragma unroll
        for (int p = 0; p < 2; ++p) {
            const int m = (2 * mp + p) * 16 + laneN;
            const unsigned int* base = (const unsigned int*)&As[m * 32];
            const int g = laneHi * 8;
#pragma unroll
            for (int d = 0; d < 4; ++d) a[p].d[d]     = base[g + d];
#pragma unroll
            for (int d = 0; d < 4; ++d) a[p].d[4 + d] = base[4 + g + d];
        }
#pragma unroll
        for (int q = 0; q < 2; ++q) {
            const int n = (2 * np + q) * 16 + laneN;
            Frag16 b;
            const unsigned int* bp = &BsU[n * 16];
            const int g = laneHi * 8;
#pragma unroll
            for (int d = 0; d < 8; ++d) b.d[d] = bp[g + d];
#pragma unroll
            for (int p = 0; p < 2; ++p)
                acc[p][q] = __builtin_amdgcn_wmma_f32_16x16x32_bf16(
                    false, a[p].v, false, b.v, (short)0, acc[p][q], false, false);
        }
    }

#pragma unroll
    for (int p = 0; p < 2; ++p) {
#pragma unroll
        for (int q = 0; q < 2; ++q) {
            const int h = h0 + (2 * np + q) * 16 + laneN;
#pragma unroll
            for (int r = 0; r < 8; ++r) {
                const int t = t0 + (2 * mp + p) * 16 + laneHi * 8 + r;
                out[(size_t)t * H_DIM + h] = acc[p][q][r];
            }
        }
    }
}

// ---------------------------------------------------------------------------
// Launch: cvt -> router -> gate_up/SiLU -> down. Stream-ordered, capture-safe.
// d_out = [ final T*H fp32 | router_logits T*E fp32 ]
// d_ws  = [ combine T*E fp32 (64KB) | xbf T*H bf16 (8MB) | HS T*8192 bf16 (32MB) ]
// ---------------------------------------------------------------------------
extern "C" void kernel_launch(void* const* d_in, const int* in_sizes, int n_in,
                              void* d_out, int out_size, void* d_ws, size_t ws_size,
                              hipStream_t stream) {
    (void)in_sizes; (void)n_in; (void)out_size; (void)ws_size;
    const float* x    = (const float*)d_in[0];
    const float* gw   = (const float*)d_in[1];
    const float* bias = (const float*)d_in[2];
    const float* gup  = (const float*)d_in[3];
    const float* wd   = (const float*)d_in[4];

    float* out    = (float*)d_out;
    float* logits = out + (size_t)T_TOK * H_DIM;

    char* ws = (char*)d_ws;
    float*  combine = (float*)ws;
    __bf16* xbf     = (__bf16*)(ws + (size_t)T_TOK * E_NUM * sizeof(float));
    __bf16* hsbuf   = (__bf16*)(ws + (size_t)T_TOK * E_NUM * sizeof(float)
                                   + (size_t)T_TOK * H_DIM * sizeof(__bf16));

    moe_cvt_x_kernel<<<(T_TOK * H_DIM) / (256 * 8), 256, 0, stream>>>(x, xbf);
    moe_router_kernel<<<T_TOK / 8, 256, 0, stream>>>(x, gw, bias, logits, combine);
    moe_gateup_kernel<<<dim3(I_DIM / 64, T_TOK / 128, E_NUM), 256, 0, stream>>>(
        xbf, gup, combine, hsbuf);
    moe_down_kernel<<<dim3(H_DIM / 64, T_TOK / 128), 256, 0, stream>>>(
        hsbuf, wd, out);
}
